// KCNetSegment_40759239639789
// MI455X (gfx1250) — matrix operand order repaired and checked by reading, not compiled, not run on hardware
//
#include <hip/hip_runtime.h>
#include <hip/hip_bf16.h>

// ---------------------------------------------------------------------------
// KCNet segmentation forward, CDNA5 (gfx1250).
// GEMMs run on v_wmma_f32_16x16x32_f16 (wave32). Each wave computes a 64x16
// C strip (4 stacked 16x16 tiles). Activations are pre-packed per K-block so
// both A and B fragments load as global_load_b128 (no scalar u16 gathers).
// ---------------------------------------------------------------------------

typedef __attribute__((ext_vector_type(16))) _Float16 v16h;
typedef __attribute__((ext_vector_type(8)))  _Float16 v8h;
typedef __attribute__((ext_vector_type(8)))  float    v8f;

#define BB   8
#define NP   2048
#define NN   16384          // BB*NP, 2^14
#define KNN  18
#define KP1  19
#define LK   16
#define MK   18
#define INV_SIGMA 200.0f    // 1/0.005
#define BN_EPS    1e-5f

// concat-buffer row offsets (channels)
#define R_X1 0
#define R_KC 3
#define R_X2 19
#define R_X3 83
#define R_X4 147
#define R_X5 275
#define R_X6 403
#define R_X7 915
#define R_OH 1939
#define C_CAT 1955
#define C_PAD 1984          // C_CAT rounded up to 32 (mlp7a Ipad)

// ---------------------------------------------------------------------------
// kNN: one thread per point, batch points staged in LDS, register top-19.
// ---------------------------------------------------------------------------
__global__ void __launch_bounds__(256)
knn_kernel(const float* __restrict__ pts, int* __restrict__ idx)
{
    __shared__ float sx[NP], sy[NP], sz[NP];
    const int b = blockIdx.y;
    const int i = blockIdx.x * 256 + threadIdx.x;
    const float* pb = pts + (size_t)b * 3 * NP;
    for (int j = threadIdx.x; j < NP; j += 256) {
        sx[j] = pb[j]; sy[j] = pb[NP + j]; sz[j] = pb[2 * NP + j];
    }
    __syncthreads();

    const float xi = sx[i], yi = sy[i], zi = sz[i];
    float bd[KP1]; int bix[KP1];
#pragma unroll
    for (int t = 0; t < KP1; ++t) { bd[t] = 3.4e38f; bix[t] = 0; }

    for (int j = 0; j < NP; ++j) {
        const float dx = sx[j] - xi, dy = sy[j] - yi, dz = sz[j] - zi;
        float d = dx * dx + dy * dy + dz * dz;
        int   id = j;
        if (d < bd[KP1 - 1]) {
#pragma unroll
            for (int p = 0; p < KP1; ++p) {
                const bool sm = d < bd[p];
                const float td = bd[p]; const int ti = bix[p];
                if (sm) { bd[p] = d; bix[p] = id; d = td; id = ti; }
            }
        }
    }
    int* op = idx + ((size_t)b * NP + i) * KP1;
#pragma unroll
    for (int t = 0; t < KP1; ++t) op[t] = bix[t];
}

// ---------------------------------------------------------------------------
// Kernel correlation: 16 threads (one per l) share a point; kernels in LDS.
// kc[b,l,n] = (1/18) * sum_k sum_m exp(-|rel_k - ker_lm|^2 / sigma)
// ---------------------------------------------------------------------------
__global__ void __launch_bounds__(256)
kc_kernel(const float* __restrict__ pts, const int* __restrict__ idx,
          const float* __restrict__ kern, _Float16* __restrict__ out /* bigcat+R_KC*NN */)
{
    __shared__ float kx[LK * MK], ky[LK * MK], kz[LK * MK], ksq[LK * MK];
    for (int t = threadIdx.x; t < LK * MK; t += 256) {
        const float a = kern[t * 3], b = kern[t * 3 + 1], c = kern[t * 3 + 2];
        kx[t] = a; ky[t] = b; kz[t] = c; ksq[t] = a * a + b * b + c * c;
    }
    __syncthreads();

    const int l = threadIdx.x & 15;
    const int p = blockIdx.x * 16 + (threadIdx.x >> 4);   // 0..NN-1
    const int b = p >> 11, n = p & (NP - 1);
    const float* pb = pts + (size_t)b * 3 * NP;
    const float x0 = pb[n], y0 = pb[NP + n], z0 = pb[2 * NP + n];
    const int* ip = idx + (size_t)p * KP1 + 1;            // skip self

    float acc = 0.0f;
    for (int k = 0; k < KNN; ++k) {
        const int jn = ip[k];
        const float rx = pb[jn] - x0;
        const float ry = pb[NP + jn] - y0;
        const float rz = pb[2 * NP + jn] - z0;
        const float rsq = rx * rx + ry * ry + rz * rz;
#pragma unroll
        for (int m = 0; m < MK; ++m) {
            const int t = l * MK + m;
            const float d2 = rsq + ksq[t] - 2.0f * (rx * kx[t] + ry * ky[t] + rz * kz[t]);
            acc += __expf(-d2 * INV_SIGMA);
        }
    }
    out[(size_t)l * NN + p] = (_Float16)(acc * (1.0f / 18.0f));
}

// ---------------------------------------------------------------------------
// small elementwise helpers
// ---------------------------------------------------------------------------
__global__ void points_to_cat(const float* __restrict__ pts, _Float16* __restrict__ out)
{
    const int t = blockIdx.x * 256 + threadIdx.x;     // 3*NN
    const int c = t >> 14, col = t & (NN - 1);
    const int b = col >> 11, n = col & (NP - 1);
    out[t] = (_Float16)pts[((size_t)b * 3 + c) * NP + n];
}

__global__ void onehot_kernel(const int* __restrict__ labels, _Float16* __restrict__ out)
{
    const int t = blockIdx.x * 256 + threadIdx.x;     // 16*NN
    const int c = t >> 14, col = t & (NN - 1);
    const int b = col >> 11;
    out[t] = (_Float16)((labels[b] == c) ? 1.0f : 0.0f);
}

__global__ void zero_h_kernel(_Float16* __restrict__ p)
{
    p[(size_t)blockIdx.x * 256 + threadIdx.x] = (_Float16)0.0f;
}

__global__ void wconv_kernel(const float* __restrict__ W, _Float16* __restrict__ Wp,
                             int O, int I, int Ipad)
{
    const int t = blockIdx.x * 256 + threadIdx.x;     // Opad*Ipad
    const int o = t / Ipad, i = t - o * Ipad;
    const float v = (o < O && i < I) ? W[(size_t)o * I + i] : 0.0f;
    Wp[t] = (_Float16)v;
}

__global__ void bpad_kernel(const float* __restrict__ b, float* __restrict__ bp, int O)
{
    const int t = blockIdx.x * 64 + threadIdx.x;      // Opad
    bp[t] = (t < O) ? b[t] : 0.0f;
}

// ---------------------------------------------------------------------------
// Pack activations X (Ipad x NN, row-major f16) into the WMMA B-fragment
// layout: out[(kb*NN + col)*32 + kk] = X[(kb*32 + kk)*NN + col].
// After packing, a lane's 16 B values (K = half*16 + j) are two b128 loads.
// LDS 32x65 tile: coalesced reads AND coalesced b128 writes.
// ---------------------------------------------------------------------------
__global__ void __launch_bounds__(256)
pack_b_kernel(const _Float16* __restrict__ in, _Float16* __restrict__ out)
{
    __shared__ _Float16 tile[32][65];
    const int kb = blockIdx.x;             // K block (32 rows)
    const int c0 = blockIdx.y * 64;        // column block
    const int r = threadIdx.x >> 6;        // 0..3
    const int c = threadIdx.x & 63;        // 0..63
#pragma unroll
    for (int s = 0; s < 8; ++s) {
        const int k = r + s * 4;
        tile[k][c] = in[((size_t)kb * 32 + k) * NN + (c0 + c)];
    }
    __syncthreads();
    const int t   = threadIdx.x;
    const int oc  = t >> 2;                // output column within block: 0..63
    const int kk0 = (t & 3) * 8;           // 0,8,16,24
    v8h v;
#pragma unroll
    for (int u = 0; u < 8; ++u) v[u] = tile[kk0 + u][oc];
    *(v8h*)(out + ((size_t)kb * NN + c0) * 32 + (size_t)t * 8) = v;
}

// ---------------------------------------------------------------------------
// GEMM: Y[o, col] = sum_i W[o,i] * Xp[i, col] + biasp[o]
// A = weights (Opad x Ipad f16, zero-padded).  Xp = packed activations.
// One wave computes a 64x16 strip: one B fragment per K-step feeds 4 WMMAs.
// ---------------------------------------------------------------------------
__global__ void __launch_bounds__(128)
gemm_wmma(const _Float16* __restrict__ Wp, const _Float16* __restrict__ Xp,
          const float* __restrict__ biasp, float* __restrict__ Y, int Ipad)
{
    const int lane = threadIdx.x & 31;
    const int wave = threadIdx.x >> 5;
    const int half = lane >> 4;
    const int id16 = lane & 15;
    const int n0 = (blockIdx.x * 4 + wave) * 16;
    const int o0 = blockIdx.y * 64;

    v8f acc0 = {}, acc1 = {}, acc2 = {}, acc3 = {};
    // B packed base: column (n0+id16), this lane's K sub-block (half*16)
    const _Float16* xpb   = Xp + (size_t)(n0 + id16) * 32 + half * 16;
    const _Float16* wbase = Wp + (size_t)(o0 + id16) * Ipad;   // A row M = id16

    for (int ks = 0; ks < Ipad; ks += 32) {
        // B K order used here (contiguous, matches A): lanes 0-15 hold K=0..15, lanes 16-31 hold K=16..31
        const _Float16* xk = xpb + (size_t)(ks >> 5) * NN * 32;
        v8h blo = *(const v8h*)(xk);
        v8h bhi = *(const v8h*)(xk + 8);
        v16h bm;
#pragma unroll
        for (int j = 0; j < 8; ++j) { bm[j] = blo[j]; bm[8 + j] = bhi[j]; }
        __builtin_prefetch(xk + (size_t)NN * 32, 0, 1);

        // A K order used here (contiguous, matches B): lanes 0-15 -> K 0..15; lanes 16-31 -> K 16..31
#pragma unroll
        for (int t = 0; t < 4; ++t) {
            const _Float16* wrow = wbase + (size_t)t * 16 * Ipad;
            v8h alo = *(const v8h*)(wrow + ks + half * 16);
            v8h ahi = *(const v8h*)(wrow + ks + half * 16 + 8);
            v16h a;
#pragma unroll
            for (int j = 0; j < 8; ++j) { a[j] = alo[j]; a[8 + j] = ahi[j]; }
            v8f* accp = (t == 0) ? &acc0 : (t == 1) ? &acc1 : (t == 2) ? &acc2 : &acc3;
            *accp = __builtin_amdgcn_wmma_f32_16x16x32_f16(
                false, a, false, bm, (short)0, *accp, false, false);
        }
    }

    // C/D 16x16 f32: VGPR r -> M = half*8 + r, N = id16. Unguarded (padded Y).
#pragma unroll
    for (int t = 0; t < 4; ++t) {
        const v8f acc = (t == 0) ? acc0 : (t == 1) ? acc1 : (t == 2) ? acc2 : acc3;
#pragma unroll
        for (int r = 0; r < 8; ++r) {
            const int o = o0 + t * 16 + half * 8 + r;
            Y[(size_t)o * NN + (n0 + id16)] = acc[r] + biasp[o];
        }
    }
}

// ---------------------------------------------------------------------------
// BatchNorm stats (mean over batch & N per channel) and normalize+ReLU -> f16
// ---------------------------------------------------------------------------
__global__ void __launch_bounds__(256)
bn_stats(const float* __restrict__ Y, float* __restrict__ mean, float* __restrict__ rstd)
{
    const int o = blockIdx.x;
    __shared__ float s1[256], s2[256];
    float a = 0.0f, b = 0.0f;
    for (int i = threadIdx.x; i < NN; i += 256) {
        const float v = Y[(size_t)o * NN + i];
        a += v; b += v * v;
    }
    s1[threadIdx.x] = a; s2[threadIdx.x] = b;
    __syncthreads();
    for (int s = 128; s > 0; s >>= 1) {
        if (threadIdx.x < s) { s1[threadIdx.x] += s1[threadIdx.x + s]; s2[threadIdx.x] += s2[threadIdx.x + s]; }
        __syncthreads();
    }
    if (threadIdx.x == 0) {
        const float m = s1[0] * (1.0f / NN);
        const float v = s2[0] * (1.0f / NN) - m * m;
        mean[o] = m;
        rstd[o] = rsqrtf(v + BN_EPS);
    }
}

__global__ void __launch_bounds__(256)
bn_relu(const float* __restrict__ Y, const float* __restrict__ mean,
        const float* __restrict__ rstd, const float* __restrict__ g,
        const float* __restrict__ be, _Float16* __restrict__ out)
{
    const size_t t = (size_t)blockIdx.x * 256 + threadIdx.x; // O*NN
    const int o = (int)(t >> 14);
    const float y = Y[t];
    const float v = (y - mean[o]) * rstd[o] * g[o] + be[o];
    out[t] = (_Float16)fmaxf(v, 0.0f);
}

// ---------------------------------------------------------------------------
// Graph max pool over the 19 kNN indices (per batch), f16 in/out.
// ---------------------------------------------------------------------------
__global__ void __launch_bounds__(256)
graph_pool(const _Float16* __restrict__ in, const int* __restrict__ idx,
           _Float16* __restrict__ out)
{
    const size_t t = (size_t)blockIdx.x * 256 + threadIdx.x;  // C*NN
    const int col = (int)(t & (NN - 1));
    const size_t c = t >> 14;
    const int b = col >> 11, n = col & (NP - 1);
    const int* ip = idx + ((size_t)b * NP + n) * KP1;
    const _Float16* row = in + c * NN + (size_t)b * NP;
    float mx = -3.4e38f;
#pragma unroll
    for (int j = 0; j < KP1; ++j) mx = fmaxf(mx, (float)row[ip[j]]);
    out[t] = (_Float16)mx;
}

// global max over N per (channel,batch), broadcast back in place
__global__ void __launch_bounds__(256)
global_max_bcast(_Float16* __restrict__ buf /* bigcat+R_X7*NN */)
{
    const int c = blockIdx.x, b = blockIdx.y;
    _Float16* row = buf + (size_t)c * NN + (size_t)b * NP;
    __shared__ float sm[256];
    float mx = -3.4e38f;
    for (int i = threadIdx.x; i < NP; i += 256) mx = fmaxf(mx, (float)row[i]);
    sm[threadIdx.x] = mx;
    __syncthreads();
    for (int s = 128; s > 0; s >>= 1) {
        if (threadIdx.x < s) sm[threadIdx.x] = fmaxf(sm[threadIdx.x], sm[threadIdx.x + s]);
        __syncthreads();
    }
    const _Float16 h = (_Float16)sm[0];
    for (int i = threadIdx.x; i < NP; i += 256) row[i] = h;
}

// final layout transform: y[o, b*NP+n] -> out[b, o, n]  (O=50)
__global__ void __launch_bounds__(256)
to_out_kernel(const float* __restrict__ y, float* __restrict__ out)
{
    const int t = blockIdx.x * 256 + threadIdx.x;   // 50*NN
    const int o = t >> 14, col = t & (NN - 1);
    const int b = col >> 11, n = col & (NP - 1);
    out[((size_t)b * 50 + o) * NP + n] = y[t];
}

// ---------------------------------------------------------------------------
// host launcher
// ---------------------------------------------------------------------------
static inline size_t align256(size_t x) { return (x + 255) & ~(size_t)255; }

extern "C" void kernel_launch(void* const* d_in, const int* in_sizes, int n_in,
                              void* d_out, int out_size, void* d_ws, size_t ws_size,
                              hipStream_t stream)
{
    (void)in_sizes; (void)n_in; (void)out_size; (void)ws_size;

    const float* points  = (const float*)d_in[0];
    const int*   labels  = (const int*)d_in[1];
    const float* kern    = (const float*)d_in[2];
    // mlp params: base index 3 + 4*layer, tuples (W, b, g, be); mlp7c = (W, b)
    const float* W[10]; const float* bi[10]; const float* gm[10]; const float* bt[10];
    for (int l = 0; l < 9; ++l) {
        W[l]  = (const float*)d_in[3 + 4 * l + 0];
        bi[l] = (const float*)d_in[3 + 4 * l + 1];
        gm[l] = (const float*)d_in[3 + 4 * l + 2];
        bt[l] = (const float*)d_in[3 + 4 * l + 3];
    }
    W[9]  = (const float*)d_in[35];
    bi[9] = (const float*)d_in[36];

    // layer shapes: O, I (index 8 unused so mlp7c sits at slot 9)
    const int Os[10] = { 64, 64, 128, 128, 512, 1024, 512, 256, 0, 50 };
    const int Is[10] = { 19, 64,  64, 128, 128,  512, 1955, 512, 0, 256 };

    char* ws = (char*)d_ws;
    size_t cur = 0;
    auto alloc = [&](size_t bytes) { void* p = ws + cur; cur = align256(cur + bytes); return p; };

    int*      idxb   = (int*)alloc((size_t)BB * NP * KP1 * 4);
    _Float16* Wp[10];
    int Opad[10], Ipad[10];
    for (int l = 0; l < 10; ++l) {
        if (l == 8) { Wp[l] = nullptr; Opad[l] = Ipad[l] = 0; continue; }
        Opad[l] = (Os[l] + 63) & ~63;       // multiple of 64 (4 tiles per wave)
        Ipad[l] = (Is[l] + 31) & ~31;       // multiple of 32 (WMMA K)
        Wp[l] = (_Float16*)alloc((size_t)Opad[l] * Ipad[l] * 2);
    }
    _Float16* bigcat  = (_Float16*)alloc((size_t)C_PAD * NN * 2);
    float*    ybuf    = (float*)alloc((size_t)1024 * NN * 4);
    _Float16* packbuf = (_Float16*)alloc((size_t)C_PAD * NN * 2);  // packed X
    _Float16* pbuf    = (_Float16*)alloc((size_t)512 * NN * 2);    // pools + h512
    _Float16* h256    = (_Float16*)alloc((size_t)256 * NN * 2);
    float*    meanb   = (float*)alloc(1024 * 4);
    float*    rstdb   = (float*)alloc(1024 * 4);
    float*    biasp   = (float*)alloc(1024 * 4);

    // weight conversion (f32 -> zero-padded f16), zero the bigcat K-pad rows
    for (int l = 0; l < 10; ++l) {
        if (l == 8) continue;
        const int tot = Opad[l] * Ipad[l];
        wconv_kernel<<<tot / 256, 256, 0, stream>>>(W[l], Wp[l], Os[l], Is[l], Ipad[l]);
    }
    zero_h_kernel<<<(C_PAD - C_CAT) * (NN / 256), 256, 0, stream>>>(bigcat + (size_t)C_CAT * NN);

    // kNN + features
    knn_kernel<<<dim3(NP / 256, BB), 256, 0, stream>>>(points, idxb);
    points_to_cat<<<3 * NN / 256, 256, 0, stream>>>(points, bigcat + (size_t)R_X1 * NN);
    kc_kernel<<<NN / 16, 256, 0, stream>>>(points, idxb, kern, bigcat + (size_t)R_KC * NN);
    onehot_kernel<<<16 * NN / 256, 256, 0, stream>>>(labels, bigcat + (size_t)R_OH * NN);

    auto gemm = [&](int l, const _Float16* X) {
        pack_b_kernel<<<dim3(Ipad[l] / 32, NN / 64), 256, 0, stream>>>(X, packbuf);
        bpad_kernel<<<Opad[l] / 64, 64, 0, stream>>>(bi[l], biasp, Os[l]);
        dim3 gg(NN / 64, Opad[l] / 64);
        gemm_wmma<<<gg, 128, 0, stream>>>(Wp[l], packbuf, biasp, ybuf, Ipad[l]);
    };
    auto cbr = [&](int l, const _Float16* X, _Float16* out) {
        gemm(l, X);
        bn_stats<<<Os[l], 256, 0, stream>>>(ybuf, meanb, rstdb);
        bn_relu<<<Os[l] * (NN / 256), 256, 0, stream>>>(ybuf, meanb, rstdb, gm[l], bt[l], out);
    };

    cbr(0, bigcat + (size_t)R_X1 * NN, bigcat + (size_t)R_X2 * NN);   // mlp1: 19 -> 64
    cbr(1, bigcat + (size_t)R_X2 * NN, bigcat + (size_t)R_X3 * NN);   // mlp2: 64 -> 64
    cbr(2, bigcat + (size_t)R_X3 * NN, bigcat + (size_t)R_X4 * NN);   // mlp3: 64 -> 128

    graph_pool<<<128 * (NN / 256), 256, 0, stream>>>(bigcat + (size_t)R_X4 * NN, idxb, pbuf);
    cbr(3, pbuf, bigcat + (size_t)R_X5 * NN);                          // mlp4: 128 -> 128
    cbr(4, bigcat + (size_t)R_X5 * NN, bigcat + (size_t)R_X6 * NN);   // mlp5: 128 -> 512

    graph_pool<<<512 * (NN / 256), 256, 0, stream>>>(bigcat + (size_t)R_X6 * NN, idxb, pbuf);
    cbr(5, pbuf, bigcat + (size_t)R_X7 * NN);                          // mlp6: 512 -> 1024
    global_max_bcast<<<dim3(1024, BB), 256, 0, stream>>>(bigcat + (size_t)R_X7 * NN);

    cbr(6, bigcat, pbuf);                                              // mlp7a: 1955 -> 512
    cbr(7, pbuf, h256);                                                // mlp7b: 512 -> 256

    // mlp7c: 256 -> 50, bias only (no BN/ReLU)
    gemm(9, h256);
    to_out_kernel<<<50 * (NN / 256), 256, 0, stream>>>(ybuf, (float*)d_out);
}
